// GATBlock_78331613545098
// MI455X (gfx1250) — matrix operand order repaired and checked by reading, not compiled, not run on hardware
//
#include <hip/hip_runtime.h>
#include <hip/hip_bf16.h>

#define DV   128   // D == EFF
#define HH   4     // heads
#define CC   32    // channels per head
#define FFNH 256   // FFN hidden

typedef __attribute__((ext_vector_type(16))) __bf16 v16bf;
typedef __attribute__((ext_vector_type(8)))  __bf16 v8bf;
typedef __attribute__((ext_vector_type(8)))  float  v8f;
typedef __attribute__((ext_vector_type(4))) unsigned int u32x4;
typedef __attribute__((ext_vector_type(8)))  int i32x8;
typedef __attribute__((ext_vector_type(4)))  int i32x4;

#if __has_builtin(__builtin_amdgcn_tensor_load_to_lds) && __has_builtin(__builtin_amdgcn_s_wait_tensorcnt)
#define USE_TDM 1
#else
#define USE_TDM 0
#endif

union Frag16 { v16bf v; v8bf h[2]; };

__device__ __forceinline__ unsigned short f2bf(float f) {
  unsigned int b = __float_as_uint(f);
  b += 0x7FFFu + ((b >> 16) & 1u);          // round-to-nearest-even
  return (unsigned short)(b >> 16);
}
// order-preserving float<->uint map for atomicMax on floats
__device__ __forceinline__ unsigned int ordf(float f) {
  unsigned int b = __float_as_uint(f);
  return (b & 0x80000000u) ? ~b : (b | 0x80000000u);
}
__device__ __forceinline__ float unordf(unsigned int u) {
  unsigned int b = (u & 0x80000000u) ? (u & 0x7FFFFFFFu) : ~u;
  return __uint_as_float(b);
}

// generic-pointer low 32 bits == wave-relative LDS byte offset (ISA 10.2 aperture map)
__device__ __forceinline__ unsigned int lds_off_of(const void* p) {
  return (unsigned int)(unsigned long long)p;
}

// ---- Tensor Data Mover: DMA a [16 rows x K cols] bf16 tile (rows contiguous,
// row stride = K elements) from global into LDS at lds_off. D# per ISA 8.3/8.4.
__device__ __forceinline__ void tdm_load_tile(unsigned int lds_off,
                                              const unsigned short* gsrc,
                                              int K, int Nc) {
#if USE_TDM
  unsigned long long ga = (unsigned long long)gsrc;
  u32x4 g0 = { 1u,                                    // count=1, is_restore=0
               lds_off,                               // lds_addr
               (unsigned int)(ga & 0xFFFFFFFFu),      // global_addr[31:0]
               ((unsigned int)((ga >> 32) & 0x01FFFFFFu)) | (2u << 30) }; // addr[56:32] | type=2
  i32x8 g1 = { (int)(1u << 16),                                            // data_size=1 (2 bytes)
               (int)(((unsigned int)K & 0xFFFFu) << 16),                   // tensor_dim0 lo16
               (int)((((unsigned int)K) >> 16) | (((unsigned int)Nc & 0xFFFFu) << 16)), // dim0 hi | dim1 lo
               (int)((((unsigned int)Nc) >> 16) | (((unsigned int)K) << 16)),           // dim1 hi | tile_dim0=K
               16,                                                         // tile_dim1=16, tile_dim2=0
               K,                                                          // tensor_dim0_stride lo32
               0, 0 };
  i32x4 z4 = { 0, 0, 0, 0 };
#if defined(__clang_major__) && (__clang_major__ >= 23)
  i32x8 z8 = { 0, 0, 0, 0, 0, 0, 0, 0 };
  __builtin_amdgcn_tensor_load_to_lds(g0, g1, z4, z4, z8, 0);
#else
  __builtin_amdgcn_tensor_load_to_lds(g0, g1, z4, z4, 0);
#endif
#endif
}

__device__ __forceinline__ void tdm_wait() {
#if USE_TDM
  __builtin_amdgcn_s_wait_tensorcnt(0);
#endif
}

// fallback LDS fill (only compiled when TDM builtin absent)
__device__ __forceinline__ void copy_tile_fallback(unsigned short* s,
                                                   const unsigned short* g,
                                                   int elems, int lane) {
#if !USE_TDM
  v8bf* sv = (v8bf*)s;
  const v8bf* gv = (const v8bf*)g;
  for (int i = lane; i < elems / 8; i += 32) sv[i] = gv[i];
#endif
}

// ---------------- conversion kernels ----------------
__global__ void k_f2bf(const float* __restrict__ in, unsigned short* __restrict__ out, long total) {
  long i = (long)blockIdx.x * blockDim.x + threadIdx.x;
  long stride = (long)gridDim.x * blockDim.x;
  for (; i < total; i += stride) out[i] = f2bf(in[i]);
}

// in: [K][Nc] row-major f32  ->  out: [Nc][K] bf16 (B-transposed for WMMA)
__global__ void k_transpose_f2bf(const float* __restrict__ in, unsigned short* __restrict__ out,
                                 int K, int Nc) {
  long total = (long)K * Nc;
  long i = (long)blockIdx.x * blockDim.x + threadIdx.x;
  long stride = (long)gridDim.x * blockDim.x;
  for (; i < total; i += stride) {
    int k = (int)(i / Nc), n = (int)(i % Nc);
    out[(size_t)n * K + k] = f2bf(in[i]);
  }
}

__global__ void k_sum(const float* __restrict__ in, long total, float* __restrict__ out) {
  long i = (long)blockIdx.x * blockDim.x + threadIdx.x;
  long stride = (long)gridDim.x * blockDim.x;
  float s = 0.f;
  for (; i < total; i += stride) s += in[i];
  for (int off = 16; off; off >>= 1) s += __shfl_xor(s, off, 32);
  if ((threadIdx.x & 31) == 0) atomicAdd(out, s);
}

// ---------------- WMMA fragment loaders ----------------
// A (16x32 bf16, row-major, ld=ldk): lane<16 row=lane, elems[0..7]=K0..7, [8..15]=K16..23
//                                    lane>=16 row=lane-16, elems=K8..15 / K24..31
__device__ __forceinline__ v16bf load_a(const unsigned short* base, int ldk, int row, int k0, int lane) {
  const int half = lane >> 4;
  const v8bf* p = (const v8bf*)(base + (size_t)row * ldk + k0 + half * 8);
  Frag16 f; f.h[0] = p[0]; f.h[1] = p[2];
  return f.v;
}
// B (32x16 bf16) from B-transposed tile in LDS [16 rows][ldk]: lane col = lane&15,
// lanes 0-15 hold K k0+0..7 / k0+16..23, lanes 16-31 hold k0+8..15 / k0+24..31
__device__ __forceinline__ v16bf load_b_lds(const unsigned short* sbase, int ldk, int k0, int lane) {
  const int half = lane >> 4;
  const v8bf* p = (const v8bf*)(sbase + (size_t)(lane & 15) * ldk + k0 + half * 8);
  Frag16 f; f.h[0] = p[0]; f.h[1] = p[2];
  return f.v;
}

// ---------------- fused xl/xr GEMM: [N,128] x [128,128] (x2), 32-row M tiles ----------------
__global__ __launch_bounds__(32)
void k_gemm_xlr(const unsigned short* __restrict__ xbf,
                const unsigned short* __restrict__ wlT,
                const unsigned short* __restrict__ wrT,
                const float* __restrict__ bl, const float* __restrict__ br,
                float* __restrict__ xl, float* __restrict__ xr, int nrows) {
  __shared__ unsigned short sB[2 * 16 * DV];     // 8 KB: wl tile | wr tile
  const int lane = threadIdx.x;
  const int tile = blockIdx.x;
  const int mt = tile >> 3, nt = tile & 7;       // 8 n-tiles (128/16)
  const int half = lane >> 4, q = lane & 15;

#if USE_TDM
  tdm_load_tile(lds_off_of(sB),            wlT + (size_t)nt * 16 * DV, DV, DV);
  tdm_load_tile(lds_off_of(sB + 16 * DV),  wrT + (size_t)nt * 16 * DV, DV, DV);
  tdm_wait();
#else
  copy_tile_fallback(sB,           wlT + (size_t)nt * 16 * DV, 16 * DV, lane);
  copy_tile_fallback(sB + 16 * DV, wrT + (size_t)nt * 16 * DV, 16 * DV, lane);
  __syncthreads();
#endif

  const int m0 = mt * 32;
  int r0 = m0 + q;      if (r0 >= nrows) r0 = nrows - 1;
  int r1 = m0 + 16 + q; if (r1 >= nrows) r1 = nrows - 1;
  const int col = nt * 16 + q;
  v8f al0 = {}, al1 = {}, ar0 = {}, ar1 = {};
#pragma unroll
  for (int k0 = 0; k0 < DV; k0 += 32) {
    v16bf a0 = load_a(xbf, DV, r0, k0, lane);
    v16bf a1 = load_a(xbf, DV, r1, k0, lane);
    v16bf bL = load_b_lds(sB,           DV, k0, lane);
    v16bf bR = load_b_lds(sB + 16 * DV, DV, k0, lane);
    al0 = __builtin_amdgcn_wmma_f32_16x16x32_bf16(false, a0, false, bL, (short)0, al0, false, false);
    ar0 = __builtin_amdgcn_wmma_f32_16x16x32_bf16(false, a0, false, bR, (short)0, ar0, false, false);
    al1 = __builtin_amdgcn_wmma_f32_16x16x32_bf16(false, a1, false, bL, (short)0, al1, false, false);
    ar1 = __builtin_amdgcn_wmma_f32_16x16x32_bf16(false, a1, false, bR, (short)0, ar1, false, false);
  }
  const float vbl = bl[col], vbr = br[col];
#pragma unroll
  for (int r = 0; r < 8; ++r) {
    int ma = m0 + half * 8 + r;                  // C/D layout: row = r + 8*(lane>>4)
    int mb = ma + 16;
    if (ma < nrows) {
      xl[(size_t)ma * DV + col] = al0[r] + vbl;
      xr[(size_t)ma * DV + col] = ar0[r] + vbr;
    }
    if (mb < nrows) {
      xl[(size_t)mb * DV + col] = al1[r] + vbl;
      xr[(size_t)mb * DV + col] = ar1[r] + vbr;
    }
  }
}

// ---------------- FFN1: relu([N,128] x [128,256] + b1) -> bf16, 32-row M tiles ----------------
__global__ __launch_bounds__(32)
void k_ffn1(const unsigned short* __restrict__ ybf,
            const unsigned short* __restrict__ w1T,
            const float* __restrict__ b1,
            unsigned short* __restrict__ hid, int nrows) {
  __shared__ unsigned short sB[16 * DV];         // 4 KB
  const int lane = threadIdx.x;
  const int tile = blockIdx.x;
  const int mt = tile >> 4, nt = tile & 15;      // 16 n-tiles (256/16)
  const int half = lane >> 4, q = lane & 15;

#if USE_TDM
  tdm_load_tile(lds_off_of(sB), w1T + (size_t)nt * 16 * DV, DV, FFNH);
  tdm_wait();
#else
  copy_tile_fallback(sB, w1T + (size_t)nt * 16 * DV, 16 * DV, lane);
  __syncthreads();
#endif

  const int m0 = mt * 32;
  int r0 = m0 + q;      if (r0 >= nrows) r0 = nrows - 1;
  int r1 = m0 + 16 + q; if (r1 >= nrows) r1 = nrows - 1;
  const int col = nt * 16 + q;
  v8f a0c = {}, a1c = {};
#pragma unroll
  for (int k0 = 0; k0 < DV; k0 += 32) {
    v16bf a0 = load_a(ybf, DV, r0, k0, lane);
    v16bf a1 = load_a(ybf, DV, r1, k0, lane);
    v16bf b  = load_b_lds(sB, DV, k0, lane);
    a0c = __builtin_amdgcn_wmma_f32_16x16x32_bf16(false, a0, false, b, (short)0, a0c, false, false);
    a1c = __builtin_amdgcn_wmma_f32_16x16x32_bf16(false, a1, false, b, (short)0, a1c, false, false);
  }
  const float vb = b1[col];
#pragma unroll
  for (int r = 0; r < 8; ++r) {
    int ma = m0 + half * 8 + r;
    int mb = ma + 16;
    if (ma < nrows) { float v = a0c[r] + vb; hid[(size_t)ma * FFNH + col] = f2bf(v > 0.f ? v : 0.f); }
    if (mb < nrows) { float v = a1c[r] + vb; hid[(size_t)mb * FFNH + col] = f2bf(v > 0.f ? v : 0.f); }
  }
}

// ---------------- FFN2: [N,256] x [256,128] + b2 -> f32, 32-row M tiles ----------------
__global__ __launch_bounds__(32)
void k_ffn2(const unsigned short* __restrict__ hid,
            const unsigned short* __restrict__ w2T,
            const float* __restrict__ b2,
            float* __restrict__ ffn, int nrows) {
  __shared__ unsigned short sB[16 * FFNH];       // 8 KB
  const int lane = threadIdx.x;
  const int tile = blockIdx.x;
  const int mt = tile >> 3, nt = tile & 7;
  const int half = lane >> 4, q = lane & 15;

#if USE_TDM
  tdm_load_tile(lds_off_of(sB), w2T + (size_t)nt * 16 * FFNH, FFNH, DV);
  tdm_wait();
#else
  copy_tile_fallback(sB, w2T + (size_t)nt * 16 * FFNH, 16 * FFNH, lane);
  __syncthreads();
#endif

  const int m0 = mt * 32;
  int r0 = m0 + q;      if (r0 >= nrows) r0 = nrows - 1;
  int r1 = m0 + 16 + q; if (r1 >= nrows) r1 = nrows - 1;
  const int col = nt * 16 + q;
  v8f a0c = {}, a1c = {};
#pragma unroll
  for (int k0 = 0; k0 < FFNH; k0 += 32) {
    v16bf a0 = load_a(hid, FFNH, r0, k0, lane);
    v16bf a1 = load_a(hid, FFNH, r1, k0, lane);
    v16bf b  = load_b_lds(sB, FFNH, k0, lane);
    a0c = __builtin_amdgcn_wmma_f32_16x16x32_bf16(false, a0, false, b, (short)0, a0c, false, false);
    a1c = __builtin_amdgcn_wmma_f32_16x16x32_bf16(false, a1, false, b, (short)0, a1c, false, false);
  }
  const float vb = b2[col];
#pragma unroll
  for (int r = 0; r < 8; ++r) {
    int ma = m0 + half * 8 + r;
    int mb = ma + 16;
    if (ma < nrows) ffn[(size_t)ma * DV + col] = a0c[r] + vb;
    if (mb < nrows) ffn[(size_t)mb * DV + col] = a1c[r] + vb;
  }
}

// ---------------- edge kernels (thread per (edge, head)) ----------------
__global__ void k_edge_logits(const float* __restrict__ xl, const float* __restrict__ xr,
                              const float* __restrict__ We, const float* __restrict__ att,
                              const int* __restrict__ ei, const float* __restrict__ ea,
                              const float* __restrict__ easum,
                              float* __restrict__ logits, unsigned int* __restrict__ mx,
                              int E, int Et) {
  int t = blockIdx.x * blockDim.x + threadIdx.x;
  if (t >= Et * HH) return;
  int e = t >> 2, h = t & 3;
  int s, d; float eav;
  if (e < E) { s = ei[e]; d = ei[E + e]; eav = ea[e]; }
  else       { s = d = e - E; eav = easum[0] / (float)E; }   // self-loop, mean edge attr
  const float* ps = xl + (size_t)s * DV + h * CC;
  const float* pd = xr + (size_t)d * DV + h * CC;
  const float* pw = We + h * CC;
  const float* pa = att + h * CC;
  float lg = 0.f;
#pragma unroll 8
  for (int c = 0; c < CC; ++c) {
    float m = ps[c] + pd[c] + eav * pw[c];
    m = (m > 0.f) ? m : 0.2f * m;                // leaky_relu 0.2
    lg += m * pa[c];
  }
  logits[t] = lg;
  atomicMax(&mx[(size_t)d * HH + h], ordf(lg));
}

__global__ void k_edge_exp(const int* __restrict__ ei,
                           const unsigned int* __restrict__ mx,
                           float* __restrict__ logits, float* __restrict__ den,
                           int E, int Et) {
  int t = blockIdx.x * blockDim.x + threadIdx.x;
  if (t >= Et * HH) return;
  int e = t >> 2, h = t & 3;
  int d = (e < E) ? ei[E + e] : (e - E);
  float m = unordf(mx[(size_t)d * HH + h]);
  float ex = __expf(logits[t] - m);
  logits[t] = ex;                                // overwrite in place
  atomicAdd(&den[(size_t)d * HH + h], ex);
}

__global__ void k_scatter(const int* __restrict__ ei,
                          const float* __restrict__ logits, const float* __restrict__ den,
                          const float* __restrict__ xl, float* __restrict__ outacc,
                          int E, int Et) {
  int t = blockIdx.x * blockDim.x + threadIdx.x;
  if (t >= Et * HH) return;
  int e = t >> 2, h = t & 3;
  int s, d;
  if (e < E) { s = ei[e]; d = ei[E + e]; } else { s = d = e - E; }
  const float* ps = xl + (size_t)s * DV + h * CC;
  __builtin_prefetch(ps, 0, 0);                  // global_prefetch_b8
  float alpha = logits[t] / den[(size_t)d * HH + h];
  float* po = outacc + (size_t)d * DV + h * CC;
#pragma unroll 8
  for (int c = 0; c < CC; ++c) atomicAdd(po + c, alpha * ps[c]);
}

// ---------------- node post: bias + ELU + gated residual (wave per node) ----------------
__global__ __launch_bounds__(256)
void k_node_post(const float* __restrict__ outacc, const float* __restrict__ x,
                 const float* __restrict__ bo, const float* __restrict__ wg,
                 const float* __restrict__ bg,
                 float* __restrict__ ybuf, unsigned short* __restrict__ ybf, int nn) {
  int wid = threadIdx.x >> 5, lane = threadIdx.x & 31;
  int n = blockIdx.x * 8 + wid;
  if (n >= nn) return;
  const int c0 = lane * 4;
  float hh[4]; float part = 0.f;
#pragma unroll
  for (int j = 0; j < 4; ++j) {
    float o = outacc[(size_t)n * DV + c0 + j] + bo[c0 + j];
    float e = (o > 0.f) ? o : expm1f(o);         // ELU(alpha=1)
    hh[j] = e;
    part += e * wg[c0 + j];
  }
  for (int off = 16; off; off >>= 1) part += __shfl_xor(part, off, 32);
  float g = 1.f / (1.f + __expf(-(part + bg[0])));
#pragma unroll
  for (int j = 0; j < 4; ++j) {
    float y = g * hh[j] + (1.f - g) * x[(size_t)n * DV + c0 + j];
    ybuf[(size_t)n * DV + c0 + j] = y;
    ybf [(size_t)n * DV + c0 + j] = f2bf(y);
  }
}

// ---------------- LayerNorm over y + ffn (wave per node) ----------------
__global__ __launch_bounds__(256)
void k_ln(const float* __restrict__ ybuf, const float* __restrict__ ffn,
          const float* __restrict__ gam, const float* __restrict__ bet,
          float* __restrict__ out, int nn) {
  int wid = threadIdx.x >> 5, lane = threadIdx.x & 31;
  int n = blockIdx.x * 8 + wid;
  if (n >= nn) return;
  const int c0 = lane * 4;
  float v[4]; float s = 0.f, sq = 0.f;
#pragma unroll
  for (int j = 0; j < 4; ++j) {
    float y = ybuf[(size_t)n * DV + c0 + j] + ffn[(size_t)n * DV + c0 + j];
    v[j] = y; s += y; sq += y * y;
  }
  for (int off = 16; off; off >>= 1) { s += __shfl_xor(s, off, 32); sq += __shfl_xor(sq, off, 32); }
  float mu = s * (1.f / DV);
  float var = sq * (1.f / DV) - mu * mu;
  float rs = rsqrtf(var + 1e-5f);
#pragma unroll
  for (int j = 0; j < 4; ++j)
    out[(size_t)n * DV + c0 + j] = (v[j] - mu) * rs * gam[c0 + j] + bet[c0 + j];
}

// ---------------- host launch ----------------
extern "C" void kernel_launch(void* const* d_in, const int* in_sizes, int n_in,
                              void* d_out, int out_size, void* d_ws, size_t ws_size,
                              hipStream_t stream) {
  (void)n_in; (void)out_size; (void)ws_size;
  const float* x   = (const float*)d_in[0];
  const float* ea  = (const float*)d_in[1];
  const float* Wl  = (const float*)d_in[2];
  const float* bl  = (const float*)d_in[3];
  const float* Wr  = (const float*)d_in[4];
  const float* br  = (const float*)d_in[5];
  const float* We  = (const float*)d_in[6];
  const float* att = (const float*)d_in[7];
  const float* bo  = (const float*)d_in[8];
  const float* Wg  = (const float*)d_in[9];
  const float* bg  = (const float*)d_in[10];
  const float* W1  = (const float*)d_in[11];
  const float* b1  = (const float*)d_in[12];
  const float* W2  = (const float*)d_in[13];
  const float* b2  = (const float*)d_in[14];
  const float* gam = (const float*)d_in[15];
  const float* bet = (const float*)d_in[16];
  const int*   ei  = (const int*)d_in[17];
  float* out = (float*)d_out;

  const int N  = in_sizes[0] / DV;     // 50000
  const int E  = in_sizes[1];          // 800000
  const int Et = E + N;                // with self loops

  char* ws = (char*)d_ws;
  size_t off = 0;
  auto alloc = [&](size_t bytes) -> char* {
    char* p = ws + off;
    off = (off + bytes + 255) & ~(size_t)255;
    return p;
  };
  unsigned short* xbf  = (unsigned short*)alloc((size_t)N * DV * 2);
  unsigned short* wlT  = (unsigned short*)alloc((size_t)DV * DV * 2);
  unsigned short* wrT  = (unsigned short*)alloc((size_t)DV * DV * 2);
  unsigned short* w1T  = (unsigned short*)alloc((size_t)DV * FFNH * 2);
  unsigned short* w2T  = (unsigned short*)alloc((size_t)FFNH * DV * 2);
  float*          xl   = (float*)alloc((size_t)N * DV * 4);
  float*          xr   = (float*)alloc((size_t)N * DV * 4);
  float*          lgts = (float*)alloc((size_t)Et * HH * 4);
  unsigned int*   mx   = (unsigned int*)alloc((size_t)N * HH * 4);
  float*          den  = (float*)alloc((size_t)N * HH * 4);
  float*          oacc = (float*)alloc((size_t)N * DV * 4);
  float*          ybuf = (float*)alloc((size_t)N * DV * 4);
  unsigned short* ybf  = (unsigned short*)alloc((size_t)N * DV * 2);
  unsigned short* hid  = (unsigned short*)alloc((size_t)N * FFNH * 2);
  float*          easm = (float*)alloc(256);
  float*          ffn  = xr;           // xr dead after edge_logits -> reuse for ffn output

  // deterministic init every call (graph-replay safe)
  hipMemsetAsync(mx,   0, (size_t)N * HH * 4, stream);
  hipMemsetAsync(den,  0, (size_t)N * HH * 4, stream);
  hipMemsetAsync(oacc, 0, (size_t)N * DV * 4, stream);
  hipMemsetAsync(easm, 0, 256, stream);

  // conversions
  {
    long tot = (long)N * DV;
    k_f2bf<<<(int)((tot + 255) / 256), 256, 0, stream>>>(x, xbf, tot);
    k_transpose_f2bf<<<64, 256, 0, stream>>>(Wl, wlT, DV, DV);
    k_transpose_f2bf<<<64, 256, 0, stream>>>(Wr, wrT, DV, DV);
    k_transpose_f2bf<<<128, 256, 0, stream>>>(W1, w1T, DV, FFNH);
    k_transpose_f2bf<<<128, 256, 0, stream>>>(W2, w2T, FFNH, DV);
    k_sum<<<256, 256, 0, stream>>>(ea, (long)E, easm);
  }

  const int mt32 = (N + 31) / 32;
  // xl = x@Wl + bl ; xr = x@Wr + br  (fused, WMMA bf16, B tiles via TDM->LDS)
  k_gemm_xlr<<<mt32 * 8, 32, 0, stream>>>(xbf, wlT, wrT, bl, br, xl, xr, N);

  // edge phase (segment softmax over dst)
  const int eth = Et * HH;
  const int eblocks = (eth + 255) / 256;
  k_edge_logits<<<eblocks, 256, 0, stream>>>(xl, xr, We, att, ei, ea, easm, lgts, mx, E, Et);
  k_edge_exp<<<eblocks, 256, 0, stream>>>(ei, mx, lgts, den, E, Et);
  k_scatter<<<eblocks, 256, 0, stream>>>(ei, lgts, den, xl, oacc, E, Et);

  // node post: bias, ELU, gate, residual
  k_node_post<<<(N + 7) / 8, 256, 0, stream>>>(oacc, x, bo, Wg, bg, ybuf, ybf, N);

  // FFN (WMMA bf16) + LayerNorm
  k_ffn1<<<mt32 * 16, 32, 0, stream>>>(ybf, w1T, b1, hid, N);
  k_ffn2<<<mt32 * 8, 32, 0, stream>>>(hid, w2T, b2, ffn, N);
  k_ln<<<(N + 7) / 8, 256, 0, stream>>>(ybuf, ffn, gam, bet, out, N);
}
